// QURNNCell_13365938225768
// MI455X (gfx1250) — matrix operand order repaired and checked by reading, not compiled, not run on hardware
//
#include <hip/hip_runtime.h>
#include <math.h>

#define BATCH   4096
#define IN_DIM  1024
#define HID_DIM 2048
#define KCAT    3072   // IN_DIM + HID_DIM
#define BM      128
#define BN      64
#define BK      32

typedef __attribute__((ext_vector_type(8)))  __bf16 v8bf;
typedef __attribute__((ext_vector_type(16))) __bf16 v16bf;
typedef __attribute__((ext_vector_type(8)))  float  v8f;
typedef __attribute__((ext_vector_type(4)))  int    i32x4;

__device__ __forceinline__ unsigned short f2bf(float f) {
    // round-to-nearest-even fp32 -> bf16
    unsigned u = __float_as_uint(f);
    u += 0x7FFFu + ((u >> 16) & 1u);
    return (unsigned short)(u >> 16);
}

__device__ __forceinline__ float sigmoidf_(float x) {
    return 1.0f / (1.0f + __expf(-x));
}

__device__ __forceinline__ v8f wmma_bf16(v16bf a, v16bf b, v8f c) {
    return __builtin_amdgcn_wmma_f32_16x16x32_bf16(
        false, a, false, b, (short)0, c, false, false);
}

// ---------------------------------------------------------------------------
// Async global -> LDS copy (16 bytes / lane), ASYNCcnt-tracked on gfx1250.
// Builtin signature (per hipcc diagnostic): (global i32x4*, local i32x4*, imm, imm)
// ---------------------------------------------------------------------------
#if __has_builtin(__builtin_amdgcn_global_load_async_to_lds_b128)
#define HAS_ASYNC_LDS 1
#else
#define HAS_ASYNC_LDS 0
#endif

__device__ __forceinline__ void async_copy_b128(const unsigned short* g,
                                                unsigned short* l) {
#if HAS_ASYNC_LDS
    typedef __attribute__((address_space(1))) i32x4 GT;
    typedef __attribute__((address_space(3))) i32x4 LT;
    i32x4* gg = (i32x4*)const_cast<unsigned short*>(g);  // generic bitcast
    i32x4* ll = (i32x4*)l;
    __builtin_amdgcn_global_load_async_to_lds_b128((GT*)gg, (LT*)ll, 0, 0);
#else
    *(uint4*)l = *(const uint4*)g;
#endif
}

__device__ __forceinline__ void wait_async_and_barrier() {
#if HAS_ASYNC_LDS
  #if __has_builtin(__builtin_amdgcn_s_wait_asynccnt)
    __builtin_amdgcn_s_wait_asynccnt(0);
  #else
    asm volatile("s_wait_asynccnt 0x0" ::: "memory");
  #endif
#endif
    __syncthreads();
}

// ---------------------------------------------------------------------------
// Convert 4 weight matrices [HID, KCAT] fp32 -> bf16 into ws (order Wt,Wa,Hw,Eh)
// ---------------------------------------------------------------------------
__global__ void cvt_weights_bf16(const float* __restrict__ wt,
                                 const float* __restrict__ wa,
                                 const float* __restrict__ hw,
                                 const float* __restrict__ eh,
                                 unsigned short* __restrict__ dst) {
    const size_t WELEM = (size_t)HID_DIM * KCAT;
    size_t i4 = (size_t)blockIdx.x * blockDim.x + threadIdx.x;
    size_t n4 = (4 * WELEM) / 4;
    if (i4 >= n4) return;
    size_t i = i4 * 4;
    size_t g = i / WELEM, k = i % WELEM;
    const float* src = (g == 0) ? wt : (g == 1) ? wa : (g == 2) ? hw : eh;
    float4 v = *(const float4*)(src + k);
    ushort4 u = make_ushort4(f2bf(v.x), f2bf(v.y), f2bf(v.z), f2bf(v.w));
    *(ushort4*)(dst + i) = u;
}

// ---------------------------------------------------------------------------
// Build combined1 = [x | h_cur] bf16 [BATCH, KCAT]; also seed combined2's x part
// ---------------------------------------------------------------------------
__global__ void cvt_inputs_bf16(const float* __restrict__ x,
                                const float* __restrict__ h,
                                unsigned short* __restrict__ comb1,
                                unsigned short* __restrict__ comb2) {
    size_t i4 = (size_t)blockIdx.x * blockDim.x + threadIdx.x;
    size_t n4 = ((size_t)BATCH * KCAT) / 4;
    if (i4 >= n4) return;
    size_t i = i4 * 4;
    int b = (int)(i / KCAT);
    int c = (int)(i % KCAT);
    float4 v;
    if (c < IN_DIM) v = *(const float4*)(x + (size_t)b * IN_DIM + c);
    else            v = *(const float4*)(h + (size_t)b * HID_DIM + (c - IN_DIM));
    ushort4 u = make_ushort4(f2bf(v.x), f2bf(v.y), f2bf(v.z), f2bf(v.w));
    *(ushort4*)(comb1 + i) = u;
    if (c < IN_DIM) *(ushort4*)(comb2 + i) = u;   // x block shared by second GEMM
}

// ---------------------------------------------------------------------------
// Fused GEMM of combined @ {Wt,Wa,Hw}^T + full elementwise epilogue.
// Block tile 128x64, 8 waves (4x2), wave tile 32x32, K-step 32,
// double-buffered async global->LDS staging.
// ---------------------------------------------------------------------------
__global__ __launch_bounds__(256)
void qurnn_gemm3(const unsigned short* __restrict__ comb1,   // [BATCH, KCAT] bf16
                 const unsigned short* __restrict__ wbf,     // [3][HID, KCAT] bf16
                 const float* __restrict__ wt_b,
                 const float* __restrict__ wa_b,
                 const float* __restrict__ hw_b,
                 const float* __restrict__ h_cur,
                 const float* __restrict__ excited_cell,
                 const float* __restrict__ time_cell,
                 float* __restrict__ out,                    // [3*BATCH*HID]
                 float* __restrict__ hwbuf,                  // [BATCH, HID] fp32
                 unsigned short* __restrict__ comb2) {       // [BATCH, KCAT] bf16
    __shared__ __align__(32) unsigned short lA[2][BM * BK];      // 2 x 8 KB
    __shared__ __align__(32) unsigned short lB[2][3 * BN * BK];  // 2 x 12 KB
    const size_t WELEM = (size_t)HID_DIM * KCAT;

    const int tid  = threadIdx.x;
    const int lane = tid & 31;
    const int wv   = tid >> 5;
    const int wm   = wv & 3;        // wave row 0..3  (32 rows each)
    const int wn   = wv >> 2;       // wave col 0..1  (32 cols each)
    const int mb   = blockIdx.y * BM;
    const int nb   = blockIdx.x * BN;

    const v8f vzero = {0.f, 0.f, 0.f, 0.f, 0.f, 0.f, 0.f, 0.f};
    v8f acc[2][2][3];               // [m-sub][n-sub][matrix]
#pragma unroll
    for (int s = 0; s < 2; ++s)
#pragma unroll
        for (int n = 0; n < 2; ++n)
#pragma unroll
            for (int g = 0; g < 3; ++g) acc[s][n][g] = vzero;

    // fragment indexing: contiguous k = 16*(lane/16) + s, same for A and B
    const int lrow = lane & 15;
    const int ac0  = (lane >> 4) << 4;   // A: K base 0 or 16
    const int bk0  = (lane >> 4) << 4;   // B: K base 0 or 16

    // cooperative staging: A = 512 16B-chunks (2/thread), B = 768 (3/thread)
    auto stage = [&](int buf, int k0) {
#pragma unroll
        for (int i = 0; i < 2; ++i) {
            const int c = tid + 256 * i;
            const int row = c >> 2, ko = (c & 3) * 8;
            async_copy_b128(comb1 + (size_t)(mb + row) * KCAT + k0 + ko,
                            &lA[buf][row * BK + ko]);
        }
#pragma unroll
        for (int i = 0; i < 3; ++i) {
            const int c = tid + 256 * i;
            const int r = c >> 2, ko = (c & 3) * 8;     // r: 0..191
            const int g = r >> 6, nl = r & 63;
            async_copy_b128(wbf + (size_t)g * WELEM + (size_t)(nb + nl) * KCAT + k0 + ko,
                            &lB[buf][r * BK + ko]);
        }
    };

    stage(0, 0);
    wait_async_and_barrier();

    int cur = 0;
    for (int k0 = 0; k0 < KCAT; k0 += BK) {
        const int nxt = cur ^ 1;
        if (k0 + BK < KCAT) stage(nxt, k0 + BK);   // overlap with WMMAs below

        v16bf af[2];
#pragma unroll
        for (int s = 0; s < 2; ++s) {
            const unsigned short* ap = &lA[cur][(wm * 32 + s * 16 + lrow) * BK + ac0];
            v8bf lo = *(const v8bf*)ap;
            v8bf hi = *(const v8bf*)(ap + 8);
            af[s] = __builtin_shufflevector(lo, hi, 0,1,2,3,4,5,6,7,8,9,10,11,12,13,14,15);
        }
#pragma unroll
        for (int g = 0; g < 3; ++g) {
#pragma unroll
            for (int n = 0; n < 2; ++n) {
                const unsigned short* bp =
                    &lB[cur][(g * BN + wn * 32 + n * 16 + lrow) * BK + bk0];
                v16bf bfr = *(const v16bf*)bp;
                acc[0][n][g] = wmma_bf16(af[0], bfr, acc[0][n][g]);
                acc[1][n][g] = wmma_bf16(af[1], bfr, acc[1][n][g]);
            }
        }
        wait_async_and_barrier();
        cur = nxt;
    }

    // ---- epilogue: lane owns column j(n), rows per C/D layout --------------
    const size_t BH = (size_t)BATCH * HID_DIM;
    const int mofs = (lane >> 4) << 3;   // lanes 16-31 hold M = r+8
#pragma unroll
    for (int n = 0; n < 2; ++n) {
        const int j = nb + wn * 32 + n * 16 + lrow;
        const float btv = wt_b[j], bav = wa_b[j], bhv = hw_b[j];
#pragma unroll
        for (int s = 0; s < 2; ++s) {
#pragma unroll
            for (int r = 0; r < 8; ++r) {
                const int m = mb + wm * 32 + s * 16 + r + mofs;
                const size_t o = (size_t)m * HID_DIM + j;
                float up_time = tanhf(acc[s][n][0][r] + btv);
                float t = fmaxf(time_cell[o] + up_time, 0.0f) - 1.0f;
                float exc0 = excited_cell[o];
                float time_next, energy, exc;
                if (t <= 0.0f) { time_next = 0.0f; energy = exc0; exc = 0.0f; }
                else           { time_next = t;    energy = 0.0f; exc = exc0; }
                float absorb  = sigmoidf_(acc[s][n][1][r] + bav);
                float excited = fmaxf(exc + absorb, 0.0f);
                float h_w     = sigmoidf_(acc[s][n][2][r] + bhv);
                float h_up    = h_cur[o] * energy;
                out[BH + o]     = excited;     // output 1: excited
                out[2 * BH + o] = time_next;   // output 2: time_next
                hwbuf[o]        = h_w;
                comb2[(size_t)m * KCAT + IN_DIM + j] = f2bf(h_up);
            }
        }
    }
}

// ---------------------------------------------------------------------------
// Second GEMM: [x | h_up] @ Eh^T, epilogue -> h_next. Same tiling, 1 matrix.
// ---------------------------------------------------------------------------
__global__ __launch_bounds__(256)
void qurnn_gemm_eh(const unsigned short* __restrict__ comb2,  // [BATCH, KCAT] bf16
                   const unsigned short* __restrict__ w_eh,   // [HID, KCAT] bf16
                   const float* __restrict__ eh_b,
                   const float* __restrict__ h_cur,
                   const float* __restrict__ hwbuf,
                   float* __restrict__ out) {
    __shared__ __align__(32) unsigned short lA[2][BM * BK];   // 2 x 8 KB
    __shared__ __align__(32) unsigned short lB[2][BN * BK];   // 2 x 4 KB

    const int tid  = threadIdx.x;
    const int lane = tid & 31;
    const int wv   = tid >> 5;
    const int wm   = wv & 3;
    const int wn   = wv >> 2;
    const int mb   = blockIdx.y * BM;
    const int nb   = blockIdx.x * BN;

    const v8f vzero = {0.f, 0.f, 0.f, 0.f, 0.f, 0.f, 0.f, 0.f};
    v8f acc[2][2];
#pragma unroll
    for (int s = 0; s < 2; ++s)
#pragma unroll
        for (int n = 0; n < 2; ++n) acc[s][n] = vzero;

    const int lrow = lane & 15;
    const int ac0  = (lane >> 4) << 4;
    const int bk0  = (lane >> 4) << 4;

    auto stage = [&](int buf, int k0) {
#pragma unroll
        for (int i = 0; i < 2; ++i) {
            const int c = tid + 256 * i;
            const int row = c >> 2, ko = (c & 3) * 8;
            async_copy_b128(comb2 + (size_t)(mb + row) * KCAT + k0 + ko,
                            &lA[buf][row * BK + ko]);
        }
        {   // B: 256 16B-chunks, 1/thread
            const int r = tid >> 2, ko = (tid & 3) * 8;
            async_copy_b128(w_eh + (size_t)(nb + r) * KCAT + k0 + ko,
                            &lB[buf][r * BK + ko]);
        }
    };

    stage(0, 0);
    wait_async_and_barrier();

    int cur = 0;
    for (int k0 = 0; k0 < KCAT; k0 += BK) {
        const int nxt = cur ^ 1;
        if (k0 + BK < KCAT) stage(nxt, k0 + BK);

        v16bf af[2];
#pragma unroll
        for (int s = 0; s < 2; ++s) {
            const unsigned short* ap = &lA[cur][(wm * 32 + s * 16 + lrow) * BK + ac0];
            v8bf lo = *(const v8bf*)ap;
            v8bf hi = *(const v8bf*)(ap + 8);
            af[s] = __builtin_shufflevector(lo, hi, 0,1,2,3,4,5,6,7,8,9,10,11,12,13,14,15);
        }
#pragma unroll
        for (int n = 0; n < 2; ++n) {
            const unsigned short* bp = &lB[cur][(wn * 32 + n * 16 + lrow) * BK + bk0];
            v16bf bfr = *(const v16bf*)bp;
            acc[0][n] = wmma_bf16(af[0], bfr, acc[0][n]);
            acc[1][n] = wmma_bf16(af[1], bfr, acc[1][n]);
        }
        wait_async_and_barrier();
        cur = nxt;
    }

    const int mofs = (lane >> 4) << 3;
#pragma unroll
    for (int n = 0; n < 2; ++n) {
        const int j = nb + wn * 32 + n * 16 + lrow;
        const float ebv = eh_b[j];
#pragma unroll
        for (int s = 0; s < 2; ++s) {
#pragma unroll
            for (int r = 0; r < 8; ++r) {
                const int m = mb + wm * 32 + s * 16 + r + mofs;
                const size_t o = (size_t)m * HID_DIM + j;
                float h_state = tanhf(acc[s][n][r] + ebv);
                float hwv = hwbuf[o];
                float hn  = tanhf((1.0f - hwv) * h_cur[o] + hwv * h_state);
                out[o] = hn;   // output 0: h_next
            }
        }
    }
}

// ---------------------------------------------------------------------------
extern "C" void kernel_launch(void* const* d_in, const int* in_sizes, int n_in,
                              void* d_out, int out_size, void* d_ws, size_t ws_size,
                              hipStream_t stream) {
    const float* x   = (const float*)d_in[0];
    const float* h   = (const float*)d_in[1];
    const float* exc = (const float*)d_in[2];
    const float* tc  = (const float*)d_in[3];
    const float* wtw = (const float*)d_in[4];
    const float* wtb = (const float*)d_in[5];
    const float* waw = (const float*)d_in[6];
    const float* wab = (const float*)d_in[7];
    const float* ehw = (const float*)d_in[8];
    const float* ehb = (const float*)d_in[9];
    const float* hww = (const float*)d_in[10];
    const float* hwb = (const float*)d_in[11];
    float* out = (float*)d_out;

    // workspace layout (128 MiB total):
    //   [4][HID][KCAT] bf16 weights (Wt,Wa,Hw,Eh)  48 MiB
    //   [BATCH][KCAT]  bf16 combined1              24 MiB
    //   [BATCH][KCAT]  bf16 combined2 (x | h_up)   24 MiB
    //   [BATCH][HID]   fp32 h_w                    32 MiB
    const size_t WELEM = (size_t)HID_DIM * KCAT;
    unsigned short* wbf   = (unsigned short*)d_ws;
    unsigned short* comb1 = wbf + 4 * WELEM;
    unsigned short* comb2 = comb1 + (size_t)BATCH * KCAT;
    float* hwbuf = (float*)(comb2 + (size_t)BATCH * KCAT);

    {
        size_t n4 = (4 * WELEM) / 4;
        cvt_weights_bf16<<<(unsigned)((n4 + 255) / 256), 256, 0, stream>>>(
            wtw, waw, hww, ehw, wbf);
    }
    {
        size_t n4 = ((size_t)BATCH * KCAT) / 4;
        cvt_inputs_bf16<<<(unsigned)((n4 + 255) / 256), 256, 0, stream>>>(
            x, h, comb1, comb2);
    }
    dim3 grid(HID_DIM / BN, BATCH / BM);
    qurnn_gemm3<<<grid, 256, 0, stream>>>(
        comb1, wbf, wtb, wab, hwb, h, exc, tc, out, hwbuf, comb2);
    qurnn_gemm_eh<<<grid, 256, 0, stream>>>(
        comb2, wbf + 3 * WELEM, ehb, h, hwbuf, out);
}
